// CosineGraphAttentionLayer_18339510354493
// MI455X (gfx1250) — matrix operand order repaired and checked by reading, not compiled, not run on hardware
//
#include <hip/hip_runtime.h>

// MI455X (gfx1250) flash-style masked cosine attention.
//   S^T = XjH @ XiH^T  (f16 WMMA, K=128), XiH pre-scaled by beta*log2e/|xi|
//   fixed-shift softmax: scores bounded (|cos|<=1, beta<1) => p = adj ? 2^s : 0
//   O^T += XjT @ P^T   (f16 WMMA, K=32),  O = O^T / l at the end.
// adj (268 MB) is the only big HBM stream. XjH/XjT tiles are shared by the
// 4 waves of a block via double-buffered global_load_async_to_lds_b128
// (ASYNCcnt), cutting L2 read traffic 3x vs per-wave register loads.

typedef _Float16 v16h __attribute__((ext_vector_type(16)));
typedef _Float16 v8h  __attribute__((ext_vector_type(8)));
typedef float    v8f  __attribute__((ext_vector_type(8)));
typedef float    v4f  __attribute__((ext_vector_type(4)));
typedef int      v4i  __attribute__((ext_vector_type(4)));

#define NROWS 8192
#define MCOLS 8192
#define DDIM  128
#define LOG2E 1.44269504088896340736f

// padded LDS row strides (f16): 136*2=272B (=4 banks skew), 40*2=80B (=20 banks skew)
#define HPAD 136
#define TPAD 40

__device__ __forceinline__ v16h pack16(v8h lo, v8h hi) {
    v16h r;
#pragma unroll
    for (int i = 0; i < 8; ++i) { r[i] = lo[i]; r[i + 8] = hi[i]; }
    return r;
}

// CDNA5 async copy: GLOBAL_LOAD_ASYNC_TO_LDS_B128 (tracked by ASYNCcnt).
__device__ __forceinline__ void async_b128(const void* g, void* l) {
    unsigned lo32 = (unsigned)(unsigned long long)l;   // LDS offset = low 32 bits
    asm volatile("global_load_async_to_lds_b128 %0, %1, off"
                 :: "v"(lo32), "v"(g) : "memory");
}

// ---------------- Prep: norms + f16 conversion + xj transpose ----------------
__global__ void cga_prep(const float* __restrict__ xi, const float* __restrict__ xj,
                         const float* __restrict__ beta,
                         _Float16* __restrict__ xiH, _Float16* __restrict__ xjH,
                         _Float16* __restrict__ xjT) {
    __shared__ float red[4];
    const int row  = blockIdx.x;
    const int t    = threadIdx.x;              // 0..127
    const bool isXi = row < NROWS;
    const float* src = isXi ? (xi + (size_t)row * DDIM)
                            : (xj + (size_t)(row - NROWS) * DDIM);
    float v = src[t];
    float s = v * v;
#pragma unroll
    for (int off = 16; off > 0; off >>= 1) s += __shfl_down(s, off, 32);
    if ((t & 31) == 0) red[t >> 5] = s;
    __syncthreads();
    const float dot   = red[0] + red[1] + red[2] + red[3];
    const float scale = rsqrtf(dot + 1e-12f);
    if (isXi) {
        // fold beta AND log2(e) so WMMA produces log2-domain scores
        xiH[(size_t)row * DDIM + t] = (_Float16)(v * scale * beta[0] * LOG2E);
    } else {
        const int r = row - NROWS;
        xjH[(size_t)r * DDIM + t] = (_Float16)(v * scale);
        xjT[(size_t)t * MCOLS + r] = (_Float16)v;     // raw xj, transposed
    }
}

// ---- issue one 32-col tile (XjH 32x128 + XjT 128x32) into LDS, split by wave ----
__device__ __forceinline__ void issue_tile_async(
        const _Float16* __restrict__ xjH, const _Float16* __restrict__ xjT,
        _Float16 (*dH)[HPAD], _Float16 (*dT)[TPAD], int cbn, int wv, int lane) {
#pragma unroll
    for (int q = 0; q < 4; ++q) {              // XjH: 2 rows (256B) per issue
        const int row = wv * 8 + q * 2 + (lane >> 4);
        const int c16 = (lane & 15) * 8;
        async_b128(xjH + (size_t)(cbn + row) * DDIM + c16, &dH[row][c16]);
    }
#pragma unroll
    for (int q = 0; q < 4; ++q) {              // XjT: 8 rows (64B each) per issue
        const int row = wv * 32 + q * 8 + (lane >> 2);
        const int seg = (lane & 3) * 8;
        async_b128(xjT + (size_t)row * MCOLS + cbn + seg, &dT[row][seg]);
    }
}

// ---- one 32-column flash step: S^T wmma, mask+exp2, P relayout, O^T wmma ----
__device__ __forceinline__ void tile_step(
        const _Float16 (*__restrict__ XH)[HPAD], const _Float16 (*__restrict__ XT)[TPAD],
        _Float16 (*__restrict__ myP)[TPAD],
        const v16h* __restrict__ Bk, v8f* __restrict__ acc, float& psum,
        const int* __restrict__ adjlane, int cb, int ln, int hsel) {
    // GEMM1: S^T(32x16) = XjH(32x128) @ XiH^T, A fragments from LDS
    v8f S0 = {0.f,0.f,0.f,0.f,0.f,0.f,0.f,0.f};
    v8f S1 = S0;
#pragma unroll
    for (int kc = 0; kc < 4; ++kc) {
        const _Float16* a0 = &XH[ln][kc * 32 + hsel * 8];
        v16h A0 = pack16(*(const v8h*)a0, *(const v8h*)(a0 + 16));
        S0 = __builtin_amdgcn_wmma_f32_16x16x32_f16(false, A0, false, Bk[kc],
                                                    (short)0, S0, false, false);
        const _Float16* a1 = &XH[16 + ln][kc * 32 + hsel * 8];
        v16h A1 = pack16(*(const v8h*)a1, *(const v8h*)(a1 + 16));
        S1 = __builtin_amdgcn_wmma_f32_16x16x32_f16(false, A1, false, Bk[kc],
                                                    (short)0, S1, false, false);
    }

    // adjacency mask stream (per-wave rows, contiguous 32B x2 per lane)
    const int* arow = adjlane + cb;
    v4i m0a = *(const v4i*)(arow);
    v4i m0b = *(const v4i*)(arow + 4);
    v4i m1a = *(const v4i*)(arow + 16);
    v4i m1b = *(const v4i*)(arow + 20);
    __builtin_prefetch(arow + 32, 0, 0);       // next adj tile -> global_prefetch

    // fixed-shift softmax numerator: p = adj ? 2^s : 0   (s already log2-domain)
    v8h p0, p1;
#pragma unroll
    for (int r = 0; r < 8; ++r) {
        const int a0 = (r < 4) ? m0a[r] : m0b[r - 4];
        const int a1 = (r < 4) ? m1a[r] : m1b[r - 4];
        const float pa = a0 ? exp2f(S0[r]) : 0.f;
        const float pb = a1 ? exp2f(S1[r]) : 0.f;
        psum += pa + pb;
        p0[r] = (_Float16)pa;
        p1[r] = (_Float16)pb;
    }

    // relayout P^T -> B operand via per-wave LDS tile (in-wave DS is in-order)
    *(v8h*)(&myP[ln][hsel * 8])      = p0;     // ds_store_b128
    *(v8h*)(&myP[ln][16 + hsel * 8]) = p1;     // ds_store_b128
    asm volatile("s_wait_dscnt 0" ::: "memory");
    v16h Bp = pack16(*(const v8h*)(&myP[ln][hsel * 8]), *(const v8h*)(&myP[ln][16 + hsel * 8]));
    asm volatile("" ::: "memory");

    // GEMM2: O^T(128x16) += XjT(128x32) @ P^T(32x16), A fragments from LDS
#pragma unroll
    for (int dt = 0; dt < 8; ++dt) {
        const _Float16* trow = &XT[dt * 16 + ln][hsel * 8];
        v16h A = pack16(*(const v8h*)trow, *(const v8h*)(trow + 16));
        acc[dt] = __builtin_amdgcn_wmma_f32_16x16x32_f16(false, A, false, Bp,
                                                         (short)0, acc[dt], false, false);
    }
}

// ---------------- Flash kernel: one wave owns 16 xi rows ----------------
__launch_bounds__(128, 1)
__global__ void cga_flash(const _Float16* __restrict__ xiH,
                          const _Float16* __restrict__ xjH,
                          const _Float16* __restrict__ xjT,
                          const int*      __restrict__ adj,
                          float*          __restrict__ out) {
    __shared__ __align__(16) _Float16 sXjH[2][32][HPAD];   // double-buffered tiles
    __shared__ __align__(16) _Float16 sXjT[2][128][TPAD];
    __shared__ __align__(16) _Float16 ldsP[4][16][TPAD];   // per-wave P relayout

    const int lane = threadIdx.x & 31;
    const int wv   = threadIdx.x >> 5;
    const int hsel = lane >> 4;
    const int ln   = lane & 15;
    const int rb   = (blockIdx.x * 4 + wv) * 16;           // xi row base

    // B operand of GEMM1: Xi tile (16 rows x 128), loop-invariant in registers.
    v16h Bk[4];
    {
        const _Float16* xrow = xiH + (size_t)(rb + ln) * DDIM + hsel * 8;
#pragma unroll
        for (int kc = 0; kc < 4; ++kc)
            Bk[kc] = pack16(*(const v8h*)(xrow + kc * 32), *(const v8h*)(xrow + kc * 32 + 16));
    }

    v8f acc[8];
#pragma unroll
    for (int i = 0; i < 8; ++i) acc[i] = {0.f,0.f,0.f,0.f,0.f,0.f,0.f,0.f};
    float psum = 0.f;
    const int* adjlane = adj + (size_t)(rb + ln) * MCOLS + hsel * 8;

    issue_tile_async(xjH, xjT, sXjH[0], sXjT[0], 0, wv, lane);   // prologue: buf0 <- cb=0

    for (int cb = 0; cb < MCOLS; cb += 64) {
        // ---- half A: compute buf0(cb), prefetch buf1(cb+32) ----
        __syncthreads();                                   // all readers of buf1 done
        issue_tile_async(xjH, xjT, sXjH[1], sXjT[1], cb + 32, wv, lane);
        asm volatile("s_wait_asynccnt 0x8" ::: "memory");  // oldest 8 (buf0) landed
        __syncthreads();                                   // whole buf0 visible
        tile_step(sXjH[0], sXjT[0], ldsP[wv], Bk, acc, psum, adjlane, cb, ln, hsel);

        // ---- half B: compute buf1(cb+32), prefetch buf0(cb+64, wrap-clamped) ----
        const int cbn = (cb + 64 < MCOLS) ? cb + 64 : 0;
        __syncthreads();
        issue_tile_async(xjH, xjT, sXjH[0], sXjT[0], cbn, wv, lane);
        asm volatile("s_wait_asynccnt 0x8" ::: "memory");
        __syncthreads();
        tile_step(sXjH[1], sXjT[1], ldsP[wv], Bk, acc, psum, adjlane, cb + 32, ln, hsel);
    }
    asm volatile("s_wait_asynccnt 0x0" ::: "memory");      // drain trailing prefetch

    // ---- epilogue: combine partner half-row sum once, O[xi][d] = acc / l ----
    const float lrow = psum + __shfl_xor(psum, 16, 32);
    const float inv  = 1.0f / fmaxf(lrow, 1e-30f);
#pragma unroll
    for (int dt = 0; dt < 8; ++dt) {
        float* orow = out + (size_t)(rb + ln) * DDIM + dt * 16 + hsel * 8;
        v4f o0 = { acc[dt][0] * inv, acc[dt][1] * inv, acc[dt][2] * inv, acc[dt][3] * inv };
        v4f o1 = { acc[dt][4] * inv, acc[dt][5] * inv, acc[dt][6] * inv, acc[dt][7] * inv };
        *(v4f*)(orow)     = o0;
        *(v4f*)(orow + 4) = o1;
    }
}

extern "C" void kernel_launch(void* const* d_in, const int* in_sizes, int n_in,
                              void* d_out, int out_size, void* d_ws, size_t ws_size,
                              hipStream_t stream) {
    const float* xi   = (const float*)d_in[0];
    const float* xj   = (const float*)d_in[1];
    const int*   adj  = (const int*)d_in[2];
    const float* beta = (const float*)d_in[3];
    float* out = (float*)d_out;

    _Float16* xiH = (_Float16*)d_ws;                        // 2 MB
    _Float16* xjH = xiH + (size_t)NROWS * DDIM;             // 2 MB
    _Float16* xjT = xjH + (size_t)MCOLS * DDIM;             // 2 MB (total 6 MB)

    cga_prep<<<NROWS + MCOLS, 128, 0, stream>>>(xi, xj, beta, xiH, xjH, xjT);
    cga_flash<<<NROWS / 64, 128, 0, stream>>>(xiH, xjH, xjT, adj, out);
}
